// GraphEncoder_31542239822511
// MI455X (gfx1250) — matrix operand order repaired and checked by reading, not compiled, not run on hardware
//
#include <hip/hip_runtime.h>
#include <hip/hip_bf16.h>
#include <math.h>

#define N_NODES 50000
#define N_EDGES 800000
#define F_INN   128
#define NHD     5
#define CCH     32
#define HCF     160        // NHD * CCH
#define NGR     64
#define EFULL   (N_EDGES + N_NODES)   // edges + self loops
#define NTILES  (HCF / 16) // 10 column tiles per output strip

typedef __attribute__((ext_vector_type(16))) _Float16 v16h;
typedef __attribute__((ext_vector_type(8)))  _Float16 v8h;
typedef __attribute__((ext_vector_type(8)))  float    v8f;

__device__ __forceinline__ float leakyf(float x, float s) { return x > 0.f ? x : s * x; }

// order-preserving uint encoding of float (for atomicMax on signed floats)
__device__ __forceinline__ unsigned f_enc(float f) {
    unsigned u = __float_as_uint(f);
    return (u & 0x80000000u) ? ~u : (u | 0x80000000u);
}
__device__ __forceinline__ float f_dec(unsigned e) {
    return (e & 0x80000000u) ? __uint_as_float(e & 0x7fffffffu) : __uint_as_float(~e);
}

// f32 -> f16 convert (activations for GEMM A operand)
__global__ void conv_f16(const float* __restrict__ src, _Float16* __restrict__ dst, int n) {
    int t = blockIdx.x * blockDim.x + threadIdx.x;
    if (t < n) dst[t] = (_Float16)src[t];
}

// Pack weights W[K][160] f32 into WMMA-B fragment order (f16):
// linear index = (((kb32*NTILES + nt)*32 + lane)*16 + j), element = W[kb32*32 + 8*half + (j&7) + 16*(j>>3)][nt*16 + l15]
__global__ void pack_w(const float* __restrict__ W, _Float16* __restrict__ Wp, int K) {
    int t = blockIdx.x * blockDim.x + threadIdx.x;
    if (t >= K * HCF) return;
    int j    = t & 15;
    int lane = (t >> 4) & 31;
    int nt   = (t >> 9) % NTILES;
    int kb32 = (t >> 9) / NTILES;
    int k = kb32 * 32 + 8 * (lane >> 4) + (j & 7) + 16 * (j >> 3);
    int n = nt * 16 + (lane & 15);
    Wp[t] = (_Float16)W[(size_t)k * HCF + n];
}

// ---------------------------------------------------------------------------
// WMMA strip GEMM: C[m0:m0+16, 0:160] = A[m0:m0+16, 0:K] @ B[0:K, 0:160]
// A: row-major f16 (two b128 loads + shuffle per fragment)
// B: pre-packed fragment-order f16 (one 32B contiguous v16h per tile/lane)
// One wave per strip; 10 independent accumulators. K % 32 == 0.
// ---------------------------------------------------------------------------
template <bool BIAS, bool LEAKY>
__global__ void gemm_wmma_strip(const _Float16* __restrict__ A, const _Float16* __restrict__ Bp,
                                float* __restrict__ C, int K,
                                const float* __restrict__ bias) {
    const int lane = threadIdx.x;
    const int half = lane >> 4;
    const int l15  = lane & 15;
    const int m0   = blockIdx.x * 16;

    v8f acc[NTILES] = {};
    const _Float16* arow = A + (size_t)(m0 + l15) * K;

    for (int kb = 0; kb < K; kb += 32) {
        // A 16x32 f16 fragment: K = kb + 8*half + 0..7  and  kb + 16 + 8*half + 0..7
        v8h alo = *(const v8h*)(arow + kb + 8 * half);
        v8h ahi = *(const v8h*)(arow + kb + 16 + 8 * half);
        v16h a = __builtin_shufflevector(alo, ahi, 0, 1, 2, 3, 4, 5, 6, 7,
                                         8, 9, 10, 11, 12, 13, 14, 15);
        const _Float16* bfrag = Bp + ((size_t)(kb >> 5) * NTILES * 32 + lane) * 16;
#pragma unroll
        for (int nt = 0; nt < NTILES; ++nt) {
            v16h b = *(const v16h*)(bfrag + (size_t)nt * 32 * 16);
            acc[nt] = __builtin_amdgcn_wmma_f32_16x16x32_f16(false, a, false, b,
                                                             (short)0, acc[nt], false, false);
        }
    }

#pragma unroll
    for (int nt = 0; nt < NTILES; ++nt) {
        int n = nt * 16 + l15;
        float bv = BIAS ? bias[n] : 0.f;
#pragma unroll
        for (int r = 0; r < 8; ++r) {
            int m = m0 + r + 8 * half;
            float v = acc[nt][r] + bv;
            if (LEAKY) v = leakyf(v, 0.01f);
            C[(size_t)m * HCF + n] = v;
        }
    }
}

// ---------------------------------------------------------------------------
// scatter-mean of edge_attr onto dst (self-loop fill_value='mean')
// ---------------------------------------------------------------------------
__global__ void edge_mean_acc(const int* __restrict__ ei, const float* __restrict__ ea,
                              float* easum, float* cnt) {
    int e = blockIdx.x * blockDim.x + threadIdx.x;
    if (e >= N_EDGES) return;
    int d = ei[N_EDGES + e];
    atomicAdd(&easum[d], ea[e]);
    atomicAdd(&cnt[d], 1.0f);
}
__global__ void edge_mean_div(const float* easum, const float* cnt, float* eam) {
    int n = blockIdx.x * blockDim.x + threadIdx.x;
    if (n < N_NODES) eam[n] = easum[n] / fmaxf(cnt[n], 1.0f);
}

// dote[h] = sum_c we[h*C+c] * ae[h*C+c]   (eh = ea*we => al_e = ea * dote)
__global__ void dote_k(const float* __restrict__ we, const float* __restrict__ ae,
                       float* dote) {
    int hd = threadIdx.x;
    if (hd < NHD) {
        float s = 0.f;
        for (int c = 0; c < CCH; ++c) s += we[hd * CCH + c] * ae[hd * CCH + c];
        dote[hd] = s;
    }
}

// per-(node,head) attention logits
__global__ void node_attn(const float* __restrict__ h, const float* __restrict__ a_s,
                          const float* __restrict__ a_d, float* alsrc, float* aldst) {
    int t = blockIdx.x * blockDim.x + threadIdx.x;
    if (t >= N_NODES * NHD) return;
    int n = t / NHD, hd = t % NHD;
    const float* hp = h + (size_t)n * HCF + hd * CCH;
    const float* sp = a_s + hd * CCH;
    const float* dp = a_d + hd * CCH;
    float s = 0.f, d = 0.f;
    for (int c = 0; c < CCH; ++c) { float v = hp[c]; s += v * sp[c]; d += v * dp[c]; }
    alsrc[t] = s;
    aldst[t] = d;
}

// pass 1: raw alpha (leaky 0.2) + segment max over dst
__global__ void edge_alpha_raw(const int* __restrict__ ei, const float* __restrict__ ea,
                               const float* __restrict__ eam, const float* __restrict__ alsrc,
                               const float* __restrict__ aldst, const float* __restrict__ dote,
                               float* araw, unsigned* amax) {
    int t = blockIdx.x * blockDim.x + threadIdx.x;
    if (t >= EFULL * NHD) return;
    int e = t / NHD, hd = t % NHD;
    int s, d; float av;
    if (e < N_EDGES) { s = ei[e]; d = ei[N_EDGES + e]; av = ea[e]; }
    else             { s = e - N_EDGES; d = s; av = eam[s]; }
    float a = alsrc[s * NHD + hd] + aldst[d * NHD + hd] + av * dote[hd];
    a = leakyf(a, 0.2f);
    araw[t] = a;
    atomicMax(&amax[d * NHD + hd], f_enc(a));
}

// pass 2: exp(a - max) + segment sum
__global__ void edge_exp(const int* __restrict__ ei, float* araw,
                         const unsigned* __restrict__ amax, float* den) {
    int t = blockIdx.x * blockDim.x + threadIdx.x;
    if (t >= EFULL * NHD) return;
    int e = t / NHD, hd = t % NHD;
    int d = (e < N_EDGES) ? ei[N_EDGES + e] : (e - N_EDGES);
    float ex = __expf(araw[t] - f_dec(amax[d * NHD + hd]));
    araw[t] = ex;
    atomicAdd(&den[d * NHD + hd], ex);
}

// pass 3: normalize (final alpha lands in d_out a1/a2 region)
__global__ void edge_norm(const int* __restrict__ ei, float* araw,
                          const float* __restrict__ den) {
    int t = blockIdx.x * blockDim.x + threadIdx.x;
    if (t >= EFULL * NHD) return;
    int e = t / NHD, hd = t % NHD;
    int d = (e < N_EDGES) ? ei[N_EDGES + e] : (e - N_EDGES);
    araw[t] = araw[t] / den[d * NHD + hd];
}

// pass 4: one wave per edge; lane covers channel, iteration == head (CCH == 32)
__global__ void edge_scatter(const int* __restrict__ ei, const float* __restrict__ alpha,
                             const float* __restrict__ h, float* acc) {
    int e = blockIdx.x * (blockDim.x >> 5) + (threadIdx.x >> 5);
    int lane = threadIdx.x & 31;
    if (e >= EFULL) return;
    int s, d;
    if (e < N_EDGES) { s = ei[e]; d = ei[N_EDGES + e]; }
    else             { s = e - N_EDGES; d = s; }
    const float* hs = h + (size_t)s * HCF;
    float* ad = acc + (size_t)d * HCF;
#pragma unroll
    for (int hd = 0; hd < NHD; ++hd) {
        float al = alpha[e * NHD + hd];          // wave-uniform
        int f = hd * CCH + lane;
        atomicAdd(&ad[f], al * hs[f]);
    }
}

// bias + leaky(0.01) + LayerNorm over 160 features, one block per node; f16 output
__global__ void finalize_ln(const float* __restrict__ acc, const float* __restrict__ bias,
                            const float* __restrict__ g, const float* __restrict__ bb,
                            _Float16* __restrict__ x) {
    __shared__ float red[256];
    int n = blockIdx.x;
    int tid = threadIdx.x;
    float v = 0.f;
    if (tid < HCF) v = leakyf(acc[(size_t)n * HCF + tid] + bias[tid], 0.01f);
    red[tid] = (tid < HCF) ? v : 0.f;
    __syncthreads();
    for (int s = 128; s > 0; s >>= 1) { if (tid < s) red[tid] += red[tid + s]; __syncthreads(); }
    float mean = red[0] / (float)HCF;
    __syncthreads();
    float dv = (tid < HCF) ? (v - mean) : 0.f;
    red[tid] = dv * dv;
    __syncthreads();
    for (int s = 128; s > 0; s >>= 1) { if (tid < s) red[tid] += red[tid + s]; __syncthreads(); }
    float var = red[0] / (float)HCF;
    if (tid < HCF)
        x[(size_t)n * HCF + tid] =
            (_Float16)((v - mean) * rsqrtf(var + 1e-5f) * g[tid] + bb[tid]);
}

// mean pool: one wave per node
__global__ void pool_acc(const float* __restrict__ x3, const int* __restrict__ batch,
                         float* pooled, float* gcnt) {
    int n = blockIdx.x * (blockDim.x >> 5) + (threadIdx.x >> 5);
    int lane = threadIdx.x & 31;
    if (n >= N_NODES) return;
    int g = batch[n];
#pragma unroll
    for (int it = 0; it < NHD; ++it) {
        int f = it * CCH + lane;
        atomicAdd(&pooled[g * HCF + f], x3[(size_t)n * HCF + f]);
    }
    if (lane == 0) atomicAdd(&gcnt[g], 1.0f);
}
__global__ void pool_div(float* pooled, const float* __restrict__ gcnt) {
    int t = blockIdx.x * blockDim.x + threadIdx.x;
    if (t >= NGR * HCF) return;
    pooled[t] /= fmaxf(gcnt[t / HCF], 1.0f);
}

// ---------------------------------------------------------------------------
extern "C" void kernel_launch(void* const* d_in, const int* in_sizes, int n_in,
                              void* d_out, int out_size, void* d_ws, size_t ws_size,
                              hipStream_t stream) {
    const float* x     = (const float*)d_in[0];
    const int*   ei    = (const int*)d_in[1];
    const float* ea    = (const float*)d_in[2];
    const int*   batch = (const int*)d_in[3];
    const float* w1  = (const float*)d_in[4];
    const float* we1 = (const float*)d_in[5];
    const float* as1 = (const float*)d_in[6];
    const float* ad1 = (const float*)d_in[7];
    const float* ae1 = (const float*)d_in[8];
    const float* b1  = (const float*)d_in[9];
    const float* w2  = (const float*)d_in[10];
    const float* we2 = (const float*)d_in[11];
    const float* as2 = (const float*)d_in[12];
    const float* ad2 = (const float*)d_in[13];
    const float* ae2 = (const float*)d_in[14];
    const float* b2  = (const float*)d_in[15];
    const float* ln1g = (const float*)d_in[16];
    const float* ln1b = (const float*)d_in[17];
    const float* ln2g = (const float*)d_in[18];
    const float* ln2b = (const float*)d_in[19];
    const float* mw  = (const float*)d_in[20];
    const float* mb  = (const float*)d_in[21];

    // output regions: x3 | pooled | a1 | a2
    float* out    = (float*)d_out;
    float* x3     = out;
    float* pooled = out + (size_t)N_NODES * HCF;
    float* a1     = pooled + (size_t)NGR * HCF;
    float* a2     = a1 + (size_t)EFULL * NHD;

    // workspace layout (float units; every chunk is a multiple of 16 floats ->
    // 64B alignment preserved for the f16 vector-load buffers)
    float* w = (float*)d_ws;
    size_t off = 0;
    float* easum = w + off; off += N_NODES;
    float* cnt   = w + off; off += N_NODES;
    float* eam   = w + off; off += N_NODES;
    float* dote  = w + off; off += 16;
    float* alsrc = w + off; off += (size_t)N_NODES * NHD;
    float* aldst = w + off; off += (size_t)N_NODES * NHD;
    float* den   = w + off; off += (size_t)N_NODES * NHD;
    unsigned* amax = (unsigned*)(w + off); off += (size_t)N_NODES * NHD;
    float* hbuf  = w + off; off += (size_t)N_NODES * HCF;
    float* accb  = w + off; off += (size_t)N_NODES * HCF;
    _Float16* xh16  = (_Float16*)(w + off); off += (size_t)N_NODES * F_INN / 2; // layer-1 A
    _Float16* xbuf16 = (_Float16*)(w + off); off += (size_t)N_NODES * HCF / 2; // x1/x2 (f16)
    _Float16* wpk   = (_Float16*)(w + off); off += (size_t)HCF * HCF / 2;      // packed weights
    float* gcnt  = w + off; off += NGR;

    const int T = 256;
    const int gemmBlocks = N_NODES / 16;      // 3125 single-wave strips
    const int wavesPerBlk = T / 32;           // 8 waves/block for wave-per-item kernels

    // ---- self-loop edge_attr mean ----
    hipMemsetAsync(easum, 0, 2 * (size_t)N_NODES * sizeof(float), stream); // easum+cnt
    edge_mean_acc<<<(N_EDGES + T - 1) / T, T, 0, stream>>>(ei, ea, easum, cnt);
    edge_mean_div<<<(N_NODES + T - 1) / T, T, 0, stream>>>(easum, cnt, eam);

    // ---- layer-1 A operand in f16 ----
    conv_f16<<<(N_NODES * F_INN + T - 1) / T, T, 0, stream>>>(x, xh16, N_NODES * F_INN);

    // ---- GAT layer runner ----
    auto run_layer = [&](const _Float16* xin, int K, const float* wmat, const float* we_,
                         const float* as_, const float* ad_, const float* ae_,
                         const float* bias, const float* lg, const float* lb,
                         float* aout, _Float16* xout) {
        pack_w<<<(K * HCF + T - 1) / T, T, 0, stream>>>(wmat, wpk, K);
        gemm_wmma_strip<false, false><<<gemmBlocks, 32, 0, stream>>>(xin, wpk, hbuf, K,
                                                                     nullptr);
        node_attn<<<(N_NODES * NHD + T - 1) / T, T, 0, stream>>>(hbuf, as_, ad_, alsrc, aldst);
        dote_k<<<1, 32, 0, stream>>>(we_, ae_, dote);
        hipMemsetAsync(den, 0, (size_t)N_NODES * NHD * sizeof(float), stream);
        hipMemsetAsync(amax, 0, (size_t)N_NODES * NHD * sizeof(unsigned), stream); // enc(min)
        hipMemsetAsync(accb, 0, (size_t)N_NODES * HCF * sizeof(float), stream);
        int eh = EFULL * NHD;
        edge_alpha_raw<<<(eh + T - 1) / T, T, 0, stream>>>(ei, ea, eam, alsrc, aldst, dote,
                                                           aout, amax);
        edge_exp<<<(eh + T - 1) / T, T, 0, stream>>>(ei, aout, amax, den);
        edge_norm<<<(eh + T - 1) / T, T, 0, stream>>>(ei, aout, den);
        edge_scatter<<<(EFULL + wavesPerBlk - 1) / wavesPerBlk, T, 0, stream>>>(ei, aout,
                                                                                hbuf, accb);
        finalize_ln<<<N_NODES, 256, 0, stream>>>(accb, bias, lg, lb, xout);
    };

    run_layer(xh16,   F_INN, w1, we1, as1, ad1, ae1, b1, ln1g, ln1b, a1, xbuf16);
    run_layer(xbuf16, HCF,   w2, we2, as2, ad2, ae2, b2, ln2g, ln2b, a2, xbuf16);

    // ---- merge MLP: x3 = leaky(x2 @ mw + mb, 0.01), straight into d_out ----
    pack_w<<<(HCF * HCF + T - 1) / T, T, 0, stream>>>(mw, wpk, HCF);
    gemm_wmma_strip<true, true><<<gemmBlocks, 32, 0, stream>>>(xbuf16, wpk, x3, HCF, mb);

    // ---- mean pool ----
    hipMemsetAsync(pooled, 0, (size_t)NGR * HCF * sizeof(float), stream);
    hipMemsetAsync(gcnt, 0, NGR * sizeof(float), stream);
    pool_acc<<<(N_NODES + wavesPerBlk - 1) / wavesPerBlk, T, 0, stream>>>(x3, batch,
                                                                          pooled, gcnt);
    pool_div<<<(NGR * HCF + T - 1) / T, T, 0, stream>>>(pooled, gcnt);
}
